// L_assign_38259568673284
// MI455X (gfx1250) — matrix operand order repaired and checked by reading, not compiled, not run on hardware
//
#include <hip/hip_runtime.h>
#include <hip/hip_bf16.h>
#include <stdint.h>

typedef __attribute__((ext_vector_type(16))) _Float16 v16h;
typedef __attribute__((ext_vector_type(8)))  float    v8f;

#define NPIX   (8 * 256 * 256)          // 524288 pixels total (all batches)
#define NBLK   (NPIX / 32)              // 16384 pixel blocks of 32
#define WGS_PER_TILE 32                 // workgroups cooperating on one 16-channel tile

// ---------------------------------------------------------------------------
// ws layout:
//   [0, 524288)            : bin index per pixel, u8
//   [524288, 524544)       : counts, u32[64]
//   [525312, 640000)       : sums, f32[448*64]  (layer offsets 0 / 4096 / 12288 floats)
// ---------------------------------------------------------------------------

__global__ __launch_bounds__(256) void init_kernel(uint32_t* counts, float* sums) {
    int i = blockIdx.x * 256 + threadIdx.x;
    if (i < 64)    counts[i] = 0u;
    if (i < 28672) sums[i]   = 0.0f;
}

// Compute depth bin per pixel (exact match of searchsorted on uniform edges),
// store u8 bin map, and histogram counts (LDS-privatized, then global atomics).
__global__ __launch_bounds__(256) void binning_kernel(const float* __restrict__ depths,
                                                      uint8_t* __restrict__ bins,
                                                      uint32_t* __restrict__ counts) {
    __shared__ uint32_t h[64];
    int t = threadIdx.x;
    if (t < 64) h[t] = 0u;
    __syncthreads();
    for (int i = blockIdx.x * 256 + t; i < NPIX; i += 512 * 256) {
        float d = depths[i];
        // 1000/64 = 15.625 is exact; division is exact for representable depths
        int b = (int)floorf(d / 15.625f);
        b = min(max(b, 0), 63);
        bins[i] = (uint8_t)b;
        atomicAdd(&h[b], 1u);
    }
    __syncthreads();
    if (t < 64) atomicAdd(&counts[t], h[t]);
}

// ---------------------------------------------------------------------------
// Per-layer bin-sum kernel: sums[c][bin] += upsampled_feature via one-hot WMMA.
//
// Tile shape per WMMA step: 32 pixels (K) x 16 channels (N), 64 bins as four
// 16-bin subtiles (M).  D = A(one-hot 16x32 f16) x B(features 32x16 f16) + C.
// Each wave keeps 4 x v8f f32 accumulators (full 64 bins x 16 channels).
// ---------------------------------------------------------------------------
__global__ __launch_bounds__(256) void layer_sums_kernel(const float* __restrict__ fmap,
                                                         int C, int h, int w,
                                                         float* __restrict__ sums,
                                                         const uint8_t* __restrict__ bins) {
    __shared__ float lacc[4 * 8 * 32];   // [sub][vgpr][lane] cross-wave reduction

    const int tid    = threadIdx.x;
    const int lane   = tid & 31;
    const int waveid = tid >> 5;
    const int hi     = lane >> 4;       // lane half: selects K range per ISA layout
    const int m      = lane & 15;       // A row (bin-in-subtile) == B column (channel)

    const int tile = blockIdx.x / WGS_PER_TILE;   // 16-channel tile index
    const int wg   = blockIdx.x % WGS_PER_TILE;
    // wave-uniform by construction; readfirstlane makes that visible to the
    // compiler so the block loop lowers to scalar control flow (EXEC stays all-1s
    // structurally, as V_WMMA requires).
    const int gw   = __builtin_amdgcn_readfirstlane(wg * 8 + waveid);
    const int hw   = h * w;
    const float sx = (float)w / 256.0f;
    const float sy = (float)h / 256.0f;

    const int c = tile * 16 + m;                  // this lane's channel (B: N = lane&15)

    v8f acc[4];
#pragma unroll
    for (int s = 0; s < 4; ++s) acc[s] = (v8f){0.f,0.f,0.f,0.f,0.f,0.f,0.f,0.f};

    for (int blk = gw; blk < NBLK; blk += WGS_PER_TILE * 8) {
        const int pixbase = blk << 5;
        const int b   = pixbase >> 16;
        const int rem = pixbase & 65535;
        const int oy  = rem >> 8;
        const int ox0 = rem & 255;

        // speculative prefetch of the next block's bin bytes (global_prefetch_b8;
        // OOB prefetch past buffer end is silently dropped per ISA)
        __builtin_prefetch(bins + ((size_t)(blk + WGS_PER_TILE * 8) << 5), 0, 0);

        // --- bins for this 32-pixel block (per-lane K set per 16-bit A layout) ---
        const uint32_t* bp32 = (const uint32_t*)(bins + pixbase);  // 32B aligned
        uint32_t blo[2], bhi[2];
        blo[0] = bp32[2 * hi + 0];      // bytes 8*hi .. 8*hi+7     -> K = e      (e<8)
        blo[1] = bp32[2 * hi + 1];
        bhi[0] = bp32[4 + 2 * hi + 0];  // bytes 16+8*hi .. 23+8*hi -> K = e+8    (e>=8)
        bhi[1] = bp32[4 + 2 * hi + 1];

        // --- vertical interpolation setup (uniform across lanes) ---
        float fy  = ((float)oy + 0.5f) * sy - 0.5f;
        float fy0 = floorf(fy);
        float wy  = fy - fy0;
        int y0 = (int)fy0;
        int y1 = y0 + 1;
        y0 = min(max(y0, 0), h - 1);
        y1 = min(max(y1, 0), h - 1);

        // --- B tile: bilinear features, element e <-> K = 8*hi + e (+8 if e>=8), N = lane&15 ---
        const float* fb = fmap + (size_t)(b * C + c) * (size_t)hw;
        const float* r0 = fb + (size_t)y0 * w;
        const float* r1 = fb + (size_t)y1 * w;
        v16h bv;
#pragma unroll
        for (int e = 0; e < 16; ++e) {
            int x = ox0 + 8 * hi + e + ((e >> 3) << 3);
            float fx  = ((float)x + 0.5f) * sx - 0.5f;
            float fx0 = floorf(fx);
            float wx  = fx - fx0;
            int x0 = (int)fx0;
            int x1 = x0 + 1;
            x0 = min(max(x0, 0), w - 1);
            x1 = min(max(x1, 0), w - 1);
            float v00 = r0[x0], v01 = r0[x1];
            float v10 = r1[x0], v11 = r1[x1];
            float top = v00 + wx * (v01 - v00);
            float bot = v10 + wx * (v11 - v10);
            bv[e] = (_Float16)(top + wy * (bot - top));
        }

        // --- 4 one-hot A subtiles + WMMA accumulate (covers bins 0..63) ---
#pragma unroll
        for (int s = 0; s < 4; ++s) {
            uint32_t target = (uint32_t)(s * 16 + m);
            v16h av;
#pragma unroll
            for (int e = 0; e < 16; ++e) {
                uint32_t byte;
                if (e < 8) byte = (blo[e >> 2] >> ((e & 3) * 8)) & 0xFFu;
                else { int e2 = e - 8; byte = (bhi[e2 >> 2] >> ((e2 & 3) * 8)) & 0xFFu; }
                av[e] = (byte == target) ? (_Float16)1.0f : (_Float16)0.0f;
            }
            acc[s] = __builtin_amdgcn_wmma_f32_16x16x32_f16(
                false, av, false, bv, (short)0, acc[s], false, false);
        }
    }

    // --- cross-wave reduction in LDS, then global atomic epilogue ---
    for (int i = tid; i < 1024; i += 256) lacc[i] = 0.0f;
    __syncthreads();
#pragma unroll
    for (int s = 0; s < 4; ++s)
#pragma unroll
        for (int r = 0; r < 8; ++r)
            atomicAdd(&lacc[(s * 8 + r) * 32 + lane], acc[s][r]);
    __syncthreads();
    for (int i = tid; i < 1024; i += 256) {
        int s = i >> 8, r = (i >> 5) & 7, L = i & 31;
        int bin = s * 16 + r + 8 * (L >> 4);        // D layout: M = r + 8*(lane>=16)
        int ch  = tile * 16 + (L & 15);             // D layout: N = lane&15
        atomicAdd(&sums[(size_t)ch * 64 + bin], lacc[i]);
    }
}

// Tiny epilogue: R = sums/counts, per-channel contrast statistic, scalar out.
__global__ __launch_bounds__(256) void finalize_kernel(const float* __restrict__ sums,
                                                       const uint32_t* __restrict__ counts,
                                                       float* __restrict__ out) {
    __shared__ float cnt[64];
    __shared__ float red[256];
    int t = threadIdx.x;
    if (t < 64) cnt[t] = (float)counts[t];
    __syncthreads();

    const int Cs[3] = {64, 128, 256};
    float partial = 0.0f;
    int off = 0;
    for (int l = 0; l < 3; ++l) {
        int C = Cs[l];
        float inv = 1.0f / (float)C;
        for (int c = t; c < C; c += 256) {
            const float* row = sums + off + (size_t)c * 64;
            int dk = (c * 64) / C;   // N_B=64, D=64; always < 64
            float rsum = 0.0f, rdk = 0.0f;
#pragma unroll
            for (int d = 0; d < 64; ++d) {
                float cn = cnt[d];
                float r = (cn > 0.0f) ? (row[d] / cn) : 0.0f;
                rsum += r;
                if (d == dk) rdk = r;
            }
            float rminus = (rsum - rdk) * (1.0f / 63.0f);
            float s = (fabsf(rdk) - fabsf(rminus)) / (fabsf(rdk) + fabsf(rminus) + 1e-6f);
            partial += s * inv;
        }
        off += C * 64;
    }
    red[t] = partial;
    __syncthreads();
    for (int st = 128; st > 0; st >>= 1) {
        if (t < st) red[t] += red[t + st];
        __syncthreads();
    }
    if (t == 0) out[0] = -(red[0] / 3.0f);   // LAMBDA=1, /L with L=3
}

extern "C" void kernel_launch(void* const* d_in, const int* in_sizes, int n_in,
                              void* d_out, int out_size, void* d_ws, size_t ws_size,
                              hipStream_t stream) {
    const float* depths = (const float*)d_in[0];
    const float* fmap0  = (const float*)d_in[1];
    const float* fmap1  = (const float*)d_in[2];
    const float* fmap2  = (const float*)d_in[3];

    uint8_t*  ws     = (uint8_t*)d_ws;
    uint8_t*  bins   = ws;
    uint32_t* counts = (uint32_t*)(ws + 524288);
    float*    sums   = (float*)(ws + 525312);

    init_kernel<<<112, 256, 0, stream>>>(counts, sums);
    binning_kernel<<<512, 256, 0, stream>>>(depths, bins, counts);

    // grid.x = (C/16) tiles * WGS_PER_TILE workgroups each
    layer_sums_kernel<<< (64 / 16) * WGS_PER_TILE, 256, 0, stream>>>(fmap0,  64, 64, 64, sums +     0, bins);
    layer_sums_kernel<<<(128 / 16) * WGS_PER_TILE, 256, 0, stream>>>(fmap1, 128, 32, 32, sums +  4096, bins);
    layer_sums_kernel<<<(256 / 16) * WGS_PER_TILE, 256, 0, stream>>>(fmap2, 256, 16, 16, sums + 12288, bins);

    finalize_kernel<<<1, 256, 0, stream>>>(sums, counts, (float*)d_out);
}
